// rate_RNN_mante_4174708212036
// MI455X (gfx1250) — matrix operand order repaired and checked by reading, not compiled, not run on hardware
//
#include <hip/hip_runtime.h>
#include <hip/hip_bf16.h>
#include <math.h>

// ---------------------------------------------------------------------------
// rate RNN scan, MI455X (gfx1250, wave32)
//   T=1000, B=64, IN=4, H=512, OUT=3, P=2 (rank-2 recurrence)
//   4 persistent workgroups x 512 threads (16 waves); 16 batches per WG.
//   Per step: one v_wmma_f32_16x16x32_bf16 per wave (fused [16x512]@[512x16]
//   projection: pout cols + Wout rows, zero padded), then each wave reduces
//   its own batch's partial column via LDS broadcast reads. 2 barriers/step.
//   Weights live in VGPRs; state (mem) lives in VGPRs; v=tanh(mem) in LDS bf16.
// ---------------------------------------------------------------------------

typedef __attribute__((ext_vector_type(16))) __bf16 v16bf;
typedef __attribute__((ext_vector_type(8)))  __bf16 v8bf;
typedef __attribute__((ext_vector_type(2)))  __bf16 v2bf;
typedef __attribute__((ext_vector_type(8)))  float  v8f;

#define T_STEPS 1000
#define BATCH   64
#define IN_DIM  4
#define HID     512
#define OUT_DIM 3
#define RANK    2
#define BT      16       // batch tile per workgroup
#define NWAVE   16       // 512 threads / wave32
#define PSTRIDE 260      // floats per batch-row of part_s (bank skew)

__device__ __forceinline__ float fast_tanh(float v) {
#if __has_builtin(__builtin_amdgcn_tanhf)
    return __builtin_amdgcn_tanhf(v);      // v_tanh_f32 (gfx1250 trans op)
#else
    return tanhf(v);
#endif
}

__device__ __forceinline__ float sum16(const float* p) {
    float a[8], b[4];
    #pragma unroll
    for (int i = 0; i < 8; ++i) a[i] = p[2*i] + p[2*i+1];
    #pragma unroll
    for (int i = 0; i < 4; ++i) b[i] = a[2*i] + a[2*i+1];
    return (b[0] + b[1]) + (b[2] + b[3]);
}

__global__ __launch_bounds__(512)
void rate_rnn_mante_kernel(const float* __restrict__ x,
                           const float* __restrict__ Win,
                           const float* __restrict__ Wout,
                           const float* __restrict__ pin,
                           const float* __restrict__ pout,
                           const float* __restrict__ l,
                           float* __restrict__ y)
{
    __shared__ __attribute__((aligned(32))) __bf16 Mcol_s[16][HID];   // 16 KB
    __shared__ __attribute__((aligned(32))) __bf16 v_s[BT][HID];      // 16 KB
    __shared__ __attribute__((aligned(16))) float  part_s[BT*PSTRIDE + 16]; // ~16.3 KB
    __shared__ float  x_s[2][BT][IN_DIM];                             // 512 B

    const int tid  = threadIdx.x;
    const int wave = tid >> 5;
    const int lane = tid & 31;
    const int b0   = blockIdx.x * BT;          // global batch offset of this WG
    const int ub   = wave;                     // batch row this wave updates
    const int ul   = lane;

    const float lm = 0.9048374180359595f;      // exp(-DT/TAUM) = exp(-0.1)
    const float om = 1.0f - lm;

    // ---- fused projection M (column-major, bf16) into LDS ------------------
    // M[h][q]: q<2 -> pout[:,q]; q in [2,5) -> Wout[q-2,:]; else 0
    for (int i = tid; i < 16 * HID; i += 512) {
        int q = i >> 9, h = i & (HID - 1);
        float v;
        if (q < RANK)                 v = pout[h * RANK + q];
        else if (q < RANK + OUT_DIM)  v = Wout[(q - RANK) * HID + h];
        else                          v = 0.0f;
        Mcol_s[q][h] = (__bf16)v;
    }

    // ---- per-thread weight slices into VGPRs (pairs of adjacent h rows) ----
    // thread owns h in {2*ul + 64*j, 2*ul + 64*j + 1}, j = 0..7, of batch ub.
    const float l0 = l[0], l1 = l[1];
    float wa[8][IN_DIM], wb[8][IN_DIM], qa[8][RANK], qb[8][RANK];
    #pragma unroll
    for (int j = 0; j < 8; ++j) {
        const int h = 2 * ul + 64 * j;
        const float4 A0 = *(const float4*)&Win[(size_t)h * IN_DIM];
        const float4 A1 = *(const float4*)&Win[(size_t)(h + 1) * IN_DIM];
        wa[j][0] = A0.x; wa[j][1] = A0.y; wa[j][2] = A0.z; wa[j][3] = A0.w;
        wb[j][0] = A1.x; wb[j][1] = A1.y; wb[j][2] = A1.z; wb[j][3] = A1.w;
        const float2 P0 = *(const float2*)&pin[(size_t)h * RANK];
        const float2 P1 = *(const float2*)&pin[(size_t)(h + 1) * RANK];
        qa[j][0] = P0.x * l0; qa[j][1] = P0.y * l1;
        qb[j][0] = P1.x * l0; qb[j][1] = P1.y * l1;
    }

    // ---- state registers; prologue v_0 = tanh(0) = 0; stage x_0 ------------
    float mem[16];
    #pragma unroll
    for (int j = 0; j < 16; ++j) mem[j] = 0.0f;
    #pragma unroll
    for (int j = 0; j < 8; ++j) {
        v2bf zz; zz[0] = (__bf16)0.0f; zz[1] = (__bf16)0.0f;
        *(v2bf*)&v_s[ub][2 * ul + 64 * j] = zz;
    }
    if (ul >= 4 && ul < 8) {
        const int ii = ul - 4;
        x_s[0][ub][ii] = x[(size_t)(b0 + ub) * IN_DIM + ii];
        __builtin_prefetch(&x[((size_t)BATCH + b0 + ub) * IN_DIM + ii], 0, 3);
    }
    __syncthreads();

    // ---- constant B fragment for this wave (its K slice [32w, 32w+32)) -----
    // 16-bit B 32x16 layout: lane = column (n = lane&15), chunks of 8 K at
    // kb and kb+16 with kb = 32w + 8*(lane>=16).
    v16bf Bf;
    {
        const int q  = lane & 15;
        const int kb = 32 * wave + ((lane >> 4) << 3);
        const v8bf blo = *(const v8bf*)&Mcol_s[q][kb];
        const v8bf bhi = *(const v8bf*)&Mcol_s[q][kb + 16];
        #pragma unroll
        for (int i = 0; i < 8; ++i) { Bf[i] = blo[i]; Bf[8 + i] = bhi[i]; }
    }
    // A fragment (16-bit A 16x32): lane = row (b = lane&15), chunks of 8 K at
    // k0 and k0+16 with k0 = 32w + 8*(lane>=16).
    const int abr = lane & 15;
    const int ak0 = 32 * wave + ((lane >> 4) << 3);
    // C/D f32 tile: VGPR j -> row j + 8*(lane>=16), col = lane&15
    const int cn = lane & 15, cmb = (lane >> 4) * 8;

    // ---- time scan: 2 barriers per step ------------------------------------
    #pragma unroll 1
    for (int t = 0; t < T_STEPS; ++t) {
        // phase B: per-wave WMMA on its K slice (EXEC all-ones here)
        {
            const v8bf lo = *(const v8bf*)&v_s[abr][ak0];
            const v8bf hi = *(const v8bf*)&v_s[abr][ak0 + 16];
            v16bf A;
            #pragma unroll
            for (int i = 0; i < 8; ++i) { A[i] = lo[i]; A[8 + i] = hi[i]; }
            v8f c = {};
            c = __builtin_amdgcn_wmma_f32_16x16x32_bf16(
                    false, A, false, Bf, (short)0, c, false, false);
            #pragma unroll
            for (int j = 0; j < 8; ++j)
                part_s[(cmb + j) * PSTRIDE + wave * 16 + cn] = c[j];
        }
        __syncthreads();

        // phase D: per-wave reduce of its batch's partials (broadcast reads),
        //          y emit (lanes 0-2), x stage (lanes 4-7), state update.
        {
            float p0[NWAVE], p1[NWAVE];
            #pragma unroll
            for (int w = 0; w < NWAVE; ++w) {
                const float2 pr = *(const float2*)&part_s[ub * PSTRIDE + w * 16];
                p0[w] = pr.x; p1[w] = pr.y;
            }
            const float s0 = sum16(p0), s1 = sum16(p1);

            if (ul < OUT_DIM && t >= 1) {               // y[t-1][b0+ub][ul]
                float po[NWAVE];
                #pragma unroll
                for (int w = 0; w < NWAVE; ++w)
                    po[w] = part_s[ub * PSTRIDE + w * 16 + RANK + ul];
                y[((size_t)(t - 1) * BATCH + b0 + ub) * OUT_DIM + ul] = sum16(po);
            }
            if (ul >= 4 && ul < 8 && t + 1 < T_STEPS) { // stage x_{t+1}
                const int ii = ul - 4;
                x_s[(t + 1) & 1][ub][ii] =
                    x[((size_t)(t + 1) * BATCH + b0 + ub) * IN_DIM + ii];
                if (t + 2 < T_STEPS)
                    __builtin_prefetch(&x[((size_t)(t + 2) * BATCH + b0 + ub) * IN_DIM + ii], 0, 3);
            }

            const float xv0 = x_s[t & 1][ub][0], xv1 = x_s[t & 1][ub][1];
            const float xv2 = x_s[t & 1][ub][2], xv3 = x_s[t & 1][ub][3];
            #pragma unroll
            for (int j = 0; j < 8; ++j) {
                const int h = 2 * ul + 64 * j;
                const float I0 = (wa[j][0] * xv0 + wa[j][1] * xv1)
                               + (wa[j][2] * xv2 + wa[j][3] * xv3)
                               + (qa[j][0] * s0 + qa[j][1] * s1);
                const float I1 = (wb[j][0] * xv0 + wb[j][1] * xv1)
                               + (wb[j][2] * xv2 + wb[j][3] * xv3)
                               + (qb[j][0] * s0 + qb[j][1] * s1);
                mem[2*j]   = lm * mem[2*j]   + om * I0;
                mem[2*j+1] = lm * mem[2*j+1] + om * I1;
                v2bf pk;
                pk[0] = (__bf16)fast_tanh(mem[2*j]);
                pk[1] = (__bf16)fast_tanh(mem[2*j+1]);
                *(v2bf*)&v_s[ub][h] = pk;               // packed b32 store
            }
        }
        __syncthreads();
    }

    // ---- epilogue: y[T-1] from v = tanh(mem_{T-1}) -------------------------
    {
        const v8bf lo = *(const v8bf*)&v_s[abr][ak0];
        const v8bf hi = *(const v8bf*)&v_s[abr][ak0 + 16];
        v16bf A;
        #pragma unroll
        for (int i = 0; i < 8; ++i) { A[i] = lo[i]; A[8 + i] = hi[i]; }
        v8f c = {};
        c = __builtin_amdgcn_wmma_f32_16x16x32_bf16(
                false, A, false, Bf, (short)0, c, false, false);
        #pragma unroll
        for (int j = 0; j < 8; ++j)
            part_s[(cmb + j) * PSTRIDE + wave * 16 + cn] = c[j];
    }
    __syncthreads();
    if (ul < OUT_DIM) {
        float po[NWAVE];
        #pragma unroll
        for (int w = 0; w < NWAVE; ++w)
            po[w] = part_s[ub * PSTRIDE + w * 16 + RANK + ul];
        y[((size_t)(T_STEPS - 1) * BATCH + b0 + ub) * OUT_DIM + ul] = sum16(po);
    }
}

extern "C" void kernel_launch(void* const* d_in, const int* in_sizes, int n_in,
                              void* d_out, int out_size, void* d_ws, size_t ws_size,
                              hipStream_t stream) {
    (void)in_sizes; (void)n_in; (void)out_size; (void)d_ws; (void)ws_size;
    const float* x    = (const float*)d_in[0];
    const float* Win  = (const float*)d_in[1];
    const float* Wout = (const float*)d_in[2];
    const float* pin  = (const float*)d_in[3];
    const float* pout = (const float*)d_in[4];
    const float* l    = (const float*)d_in[5];
    float* y = (float*)d_out;
    hipLaunchKernelGGL(rate_rnn_mante_kernel,
                       dim3(BATCH / BT), dim3(512), 0, stream,
                       x, Win, Wout, pin, pout, l, y);
}
